// LSTM_21088289423625
// MI455X (gfx1250) — matrix operand order repaired and checked by reading, not compiled, not run on hardware
//
#include <hip/hip_runtime.h>

typedef __bf16 bf16_t;
typedef __attribute__((ext_vector_type(16))) __bf16 v16bf;
typedef __attribute__((ext_vector_type(8)))  __bf16 v8bf;
typedef __attribute__((ext_vector_type(8)))  float  v8f;

#define T_STEPS 256
#define BATCH   64
#define DIN     512
#define HID     1024
#define DOUT    512
#define FOURH   (4 * HID)

// CDNA5 async global->LDS path (ASYNCcnt) when the toolchain exposes it.
// Probe-discovered prototype (from clang diagnostic):
//   void __builtin_amdgcn_global_load_async_to_lds_b128(
//       v4i32 __device__* gsrc, v4i32 __shared__* ldst, imm int, imm int)
#if __has_builtin(__builtin_amdgcn_global_load_async_to_lds_b128) && \
    __has_builtin(__builtin_amdgcn_s_wait_asynccnt)
#define USE_ASYNC_LDS 1
typedef int v4i_t __attribute__((vector_size(16)));
typedef __attribute__((address_space(1))) v4i_t* gv4i_p;  // global (AS1)
typedef __attribute__((address_space(3))) v4i_t* lv4i_p;  // LDS (AS3, 32-bit)
#define COPY16(d, s)                                                        \
  __builtin_amdgcn_global_load_async_to_lds_b128(                           \
      (gv4i_p)(uintptr_t)(s), (lv4i_p)(unsigned)(uintptr_t)(d), 0, 0)
#define WAIT_STAGE() __builtin_amdgcn_s_wait_asynccnt(0)
#else
#define USE_ASYNC_LDS 0
#define COPY16(d, s) (*(v8bf*)(d) = *(const v8bf*)(s))
#define WAIT_STAGE() ((void)0)
#endif

// ---------------------------------------------------------------------------
// fp32 -> bf16 round-to-nearest-even
// ---------------------------------------------------------------------------
__device__ __forceinline__ bf16_t f2bf(float f) {
  unsigned u = __builtin_bit_cast(unsigned, f);
  unsigned r = (u + 0x7FFFu + ((u >> 16) & 1u)) >> 16;
  return __builtin_bit_cast(bf16_t, (unsigned short)r);
}

__device__ __forceinline__ float sigf(float x) {
  return 1.f / (1.f + __expf(-x));
}

__global__ void convert_f32_bf16(const float* __restrict__ in,
                                 bf16_t* __restrict__ out, int n) {
  int i = blockIdx.x * blockDim.x + threadIdx.x;
  if (i < n) out[i] = f2bf(in[i]);
}

// Plain transpose: W [K][N] fp32 -> Wt [N][K] bf16
__global__ void transpose_convert(const float* __restrict__ W,
                                  bf16_t* __restrict__ Wt, int K, int N) {
  int i = blockIdx.x * blockDim.x + threadIdx.x;
  if (i >= K * N) return;
  int k = i / N, n = i % N;
  Wt[(size_t)n * K + k] = f2bf(W[i]);
}

// Gate-permuted transpose for LSTM weights: W [K][4H] -> Wt [n'][K], where
// natural column n = g*H + j  (g = gate i/f/c/o, j = hidden unit) maps to
//   n' = 64*(j>>4) + 16*g + (j&15)
// so every 64-column group holds 16 units x 4 gates -> one wave tile covers
// all four gates of its units with identical lane alignment.
__device__ __forceinline__ int gate_perm(int n) {
  int g = n >> 10;        // n / H
  int j = n & (HID - 1);  // n % H
  return ((j >> 4) << 6) + (g << 4) + (j & 15);
}

__global__ void transpose_convert_gates(const float* __restrict__ W,
                                        bf16_t* __restrict__ Wt, int K) {
  int i = blockIdx.x * blockDim.x + threadIdx.x;
  if (i >= K * FOURH) return;
  int k = i / FOURH, n = i % FOURH;
  Wt[(size_t)gate_perm(n) * K + k] = f2bf(W[i]);
}

__global__ void permute_bias(const float* __restrict__ b,
                             float* __restrict__ bp) {
  int n = blockIdx.x * blockDim.x + threadIdx.x;
  if (n < FOURH) bp[gate_perm(n)] = b[n];
}

// ---------------------------------------------------------------------------
// GEMM: [64][K1] * B1t^T (+ [64][K2] * B2t^T) + bias, bf16 in / f32 acc via
// V_WMMA_F32_16X16X32_BF16. 256 threads = 8 waves, block tile 64x128,
// wave tile 16x64 (4 accumulators), double-buffered LDS staging.
// FUSED=true: accumulators are (i,f,c_hat,o) of 16 units; epilogue applies
// the LSTM cell in-register and writes c (in/out), h (f32), h (bf16).
// FUSED=false: plain C = acc + bias (output projection).
// ---------------------------------------------------------------------------
template <bool FUSED>
__launch_bounds__(256, 2)
__global__ void gemm64(const bf16_t* __restrict__ A1, int K1,
                       const bf16_t* __restrict__ B1t,
                       const bf16_t* __restrict__ A2, int K2,
                       const bf16_t* __restrict__ B2t,
                       const float* __restrict__ bias,
                       float* __restrict__ C, int N,
                       float* __restrict__ cst, float* __restrict__ hst,
                       bf16_t* __restrict__ hbst) {
  // +8 bf16 pad per row; row stride 80B (16B multiple). Two buffers.
  __shared__ __attribute__((aligned(16))) bf16_t As[2][64][40];
  __shared__ __attribute__((aligned(16))) bf16_t Bs[2][128][40];

  const int tid  = threadIdx.x;
  const int lane = tid & 31;
  const int wave = tid >> 5;    // 0..7
  const int half = lane >> 4;   // 0/1 : K-half per 16-bit WMMA VGPR layout
  const int l16  = lane & 15;
  const int wm   = wave >> 1;   // 0..3 -> rows 16*wm
  const int wn   = wave & 1;    // 0..1 -> cols 64*wn
  const int n0   = blockIdx.x * 128;

  const v8f vzero = {0.f, 0.f, 0.f, 0.f, 0.f, 0.f, 0.f, 0.f};
  v8f acc[4];
#pragma unroll
  for (int i = 0; i < 4; ++i) acc[i] = vzero;

  const int nIter1 = K1 >> 5;
  const int nIter2 = (A2 != nullptr) ? (K2 >> 5) : 0;
  const int nIter  = nIter1 + nIter2;

  auto stage = [&](int it, int buf) {
    const bf16_t* A;
    const bf16_t* Bt;
    int K, kc;
    if (it < nIter1) { A = A1; Bt = B1t; K = K1; kc = it << 5; }
    else             { A = A2; Bt = B2t; K = K2; kc = (it - nIter1) << 5; }
    {  // A tile 64x32: thread -> row tid>>2, one 16B chunk
      int r = tid >> 2, cc = (tid & 3) * 8;
      const bf16_t* s = &A[(size_t)r * K + kc + cc];
      COPY16(&As[buf][r][cc], s);
      __builtin_prefetch(s + 32, 0, 1);  // global_prefetch_b8: next K chunk
    }
    {  // B tile 128x32 (Bt is [N][K]): row tid>>1, two 16B chunks
      int r = tid >> 1, cc = (tid & 1) * 16;
      const bf16_t* s = &Bt[(size_t)(n0 + r) * K + kc + cc];
      COPY16(&Bs[buf][r][cc], s);
      COPY16(&Bs[buf][r][cc + 8], s + 8);
      __builtin_prefetch(s + 32, 0, 1);
    }
  };

  stage(0, 0);
  for (int it = 0; it < nIter; ++it) {
    const int buf = it & 1;
    WAIT_STAGE();        // my async loads for tile `it` have landed in LDS
    __syncthreads();     // everyone's tile `it` visible; buf^1 free to reuse
    if (it + 1 < nIter) stage(it + 1, buf ^ 1);

    // A fragment: lanes 0-15 K {0..7,16..23}, lanes 16-31 K {8..15,24..31}
    union { v16bf v; v8bf h[2]; } aF;
    {
      int r = wm * 16 + l16;
      aF.h[0] = *(const v8bf*)&As[buf][r][half * 8];
      aF.h[1] = *(const v8bf*)&As[buf][r][16 + half * 8];
    }
#pragma unroll
    for (int i = 0; i < 4; ++i) {
      // B fragment: lanes 0-15 K {0..7,16..23}, lanes 16-31 K {8..15,24..31}
      union { v16bf v; v8bf h[2]; } bF;
      int r = wn * 64 + i * 16 + l16;
      bF.h[0] = *(const v8bf*)&Bs[buf][r][half * 8];
      bF.h[1] = *(const v8bf*)&Bs[buf][r][16 + half * 8];
      acc[i] = __builtin_amdgcn_wmma_f32_16x16x32_bf16(
          false, aF.v, false, bF.v, (short)0, acc[i], false, false);
    }
  }

  const int colBase = n0 + wn * 64;
  if constexpr (FUSED) {
    // acc[g] = gate g pre-activation for units u = (colBase/64)*16 + l16,
    // rows m = wm*16 + j + 8*half (C/D VGPR layout).
    const int u   = ((colBase >> 6) << 4) + l16;
    const float bI  = bias[colBase + 0  + l16];
    const float bF_ = bias[colBase + 16 + l16];
    const float bC  = bias[colBase + 32 + l16];
    const float bO  = bias[colBase + 48 + l16];
#pragma unroll
    for (int j = 0; j < 8; ++j) {
      int m = wm * 16 + j + 8 * half;
      size_t idx = (size_t)m * HID + u;
      float gi = acc[0][j] + bI;
      float gf = acc[1][j] + bF_;
      float gc = acc[2][j] + bC;
      float go = acc[3][j] + bO;
      float cn = sigf(gf) * cst[idx] + sigf(gi) * tanhf(gc);
      float hn = sigf(go) * tanhf(cn);
      cst[idx]  = cn;
      hst[idx]  = hn;
      hbst[idx] = f2bf(hn);
    }
  } else {
#pragma unroll
    for (int i = 0; i < 4; ++i) {
      int n = colBase + i * 16 + l16;
      float bv = bias[n];
#pragma unroll
      for (int j = 0; j < 8; ++j) {
        int m = wm * 16 + j + 8 * half;
        C[(size_t)m * N + n] = acc[i][j] + bv;
      }
    }
  }
}

// ---------------------------------------------------------------------------
extern "C" void kernel_launch(void* const* d_in, const int* in_sizes, int n_in,
                              void* d_out, int out_size, void* d_ws, size_t ws_size,
                              hipStream_t stream) {
  (void)in_sizes; (void)n_in; (void)out_size; (void)ws_size;

  const float* inputs = (const float*)d_in[0];
  const float* Wx0    = (const float*)d_in[1];
  const float* Wh0    = (const float*)d_in[2];
  const float* b0     = (const float*)d_in[3];
  const float* Wx1    = (const float*)d_in[4];
  const float* Wh1    = (const float*)d_in[5];
  const float* b1     = (const float*)d_in[6];
  const float* Wout   = (const float*)d_in[7];
  const float* bout   = (const float*)d_in[8];
  float* out = (float*)d_out;

  char* p = (char*)d_ws;
  auto carve = [&](size_t bytes) {
    char* r = p;
    p += (bytes + 255) & ~(size_t)255;
    return r;
  };

  bf16_t* Xb    = (bf16_t*)carve((size_t)T_STEPS * BATCH * DIN * 2);  // 16 MB
  bf16_t* Wx0t  = (bf16_t*)carve((size_t)FOURH * DIN * 2);            //  4 MB
  bf16_t* Wh0t  = (bf16_t*)carve((size_t)FOURH * HID * 2);            //  8 MB
  bf16_t* Wx1t  = (bf16_t*)carve((size_t)FOURH * HID * 2);            //  8 MB
  bf16_t* Wh1t  = (bf16_t*)carve((size_t)FOURH * HID * 2);            //  8 MB
  bf16_t* Woutt = (bf16_t*)carve((size_t)DOUT * HID * 2);             //  1 MB
  float*  b0p   = (float*)carve((size_t)FOURH * 4);
  float*  b1p   = (float*)carve((size_t)FOURH * 4);
  float*  h0f   = (float*)carve((size_t)BATCH * HID * 4);  // scratch
  float*  h1f   = (float*)carve((size_t)BATCH * HID * 4);  // final h
  char* stateBase = p;
  float*  c0    = (float*)carve((size_t)BATCH * HID * 4);
  float*  c1    = (float*)carve((size_t)BATCH * HID * 4);
  bf16_t* h0b[2], *h1b[2];
  h0b[0] = (bf16_t*)carve((size_t)BATCH * HID * 2);  // ping-pong: fused kernel
  h0b[1] = (bf16_t*)carve((size_t)BATCH * HID * 2);  // reads prev, writes next
  h1b[0] = (bf16_t*)carve((size_t)BATCH * HID * 2);
  h1b[1] = (bf16_t*)carve((size_t)BATCH * HID * 2);
  size_t stateBytes = (size_t)(p - stateBase);

  // zero-init recurrent state every call (graph-replay safe memset node)
  (void)hipMemsetAsync(stateBase, 0, stateBytes, stream);

  // one-time conversions (fully parallel)
  {
    int n = T_STEPS * BATCH * DIN;
    convert_f32_bf16<<<(n + 255) / 256, 256, 0, stream>>>(inputs, Xb, n);
  }
  transpose_convert_gates<<<(DIN * FOURH + 255) / 256, 256, 0, stream>>>(Wx0, Wx0t, DIN);
  transpose_convert_gates<<<(HID * FOURH + 255) / 256, 256, 0, stream>>>(Wh0, Wh0t, HID);
  transpose_convert_gates<<<(HID * FOURH + 255) / 256, 256, 0, stream>>>(Wx1, Wx1t, HID);
  transpose_convert_gates<<<(HID * FOURH + 255) / 256, 256, 0, stream>>>(Wh1, Wh1t, HID);
  transpose_convert<<<(HID * DOUT + 255) / 256, 256, 0, stream>>>(Wout, Woutt, HID, DOUT);
  permute_bias<<<FOURH / 256, 256, 0, stream>>>(b0, b0p);
  permute_bias<<<FOURH / 256, 256, 0, stream>>>(b1, b1p);

  // recurrent loop: 3 launches per step (2 fused + projection); bf16 weights (~29 MB) stay
  // L2-resident (192 MB L2), so steps 1..255 stream weights from L2.
  for (int t = 0; t < T_STEPS; ++t) {
    const int in = t & 1, ot = in ^ 1;
    // layer 0: gates = b0 + x_t*Wx0 + h0*Wh0 ; cell fused in epilogue
    gemm64<true><<<FOURH / 128, 256, 0, stream>>>(
        Xb + (size_t)t * BATCH * DIN, DIN, Wx0t, h0b[in], HID, Wh0t,
        b0p, nullptr, FOURH, c0, h0f, h0b[ot]);
    // layer 1: gates = b1 + h0*Wx1 + h1*Wh1 ; cell fused
    gemm64<true><<<FOURH / 128, 256, 0, stream>>>(
        h0b[ot], HID, Wx1t, h1b[in], HID, Wh1t,
        b1p, nullptr, FOURH, c1, h1f, h1b[ot]);
    // projection: out[t] = h1*Wout + bout
    gemm64<false><<<DOUT / 128, 256, 0, stream>>>(
        h1b[ot], HID, Woutt, nullptr, 0, nullptr,
        bout, out + (size_t)t * BATCH * DOUT, DOUT, nullptr, nullptr, nullptr);
  }

  // final (h, c) of layer 1 appended after outs
  float* tail = out + (size_t)T_STEPS * BATCH * DOUT;
  (void)hipMemcpyAsync(tail, h1f, (size_t)BATCH * HID * 4,
                       hipMemcpyDeviceToDevice, stream);
  (void)hipMemcpyAsync(tail + (size_t)BATCH * HID, c1, (size_t)BATCH * HID * 4,
                       hipMemcpyDeviceToDevice, stream);
}
